// SeqRNN_16381005267369
// MI455X (gfx1250) — compile-verified, matrix operand order checked by reading, not run on hardware
//
#include <hip/hip_runtime.h>
#include <math.h>

// ---------------------------------------------------------------------------
// CDNA5 (gfx1250, wave32) GRU LM forward:
//   embed-gather -> [per layer] big input-proj GEMM + 256 sequential
//   (GEMM_rz -> ew_rz -> GEMM_h -> ew_h) steps -> decoder GEMM.
// GEMMs: v_wmma_f32_16x16x32_bf16; A tile staged by the Tensor Data Mover
// (tensor_load_to_lds, TENSORcnt), W tile by global_load_async_to_lds_b128
// (ASYNCcnt). Decoder epilogue uses non-temporal stores.
// ---------------------------------------------------------------------------

typedef __bf16 bf16;
typedef __bf16 v16bf __attribute__((ext_vector_type(16)));
typedef float  v8f   __attribute__((ext_vector_type(8)));
typedef unsigned int su4 __attribute__((ext_vector_type(4)));
typedef unsigned int su8 __attribute__((ext_vector_type(8)));

constexpr int Bc = 16, Sc = 256, Vc = 32000, Hc = 1024, Lc = 2;

__device__ __forceinline__ float sigmoidf_(float x) {
    return 1.0f / (1.0f + __expf(-x));
}

// ---------------------------------------------------------------- converters
__global__ void cvt_f32_bf16_kernel(const float* __restrict__ src,
                                    bf16* __restrict__ dst, int n) {
    for (int i = blockIdx.x * blockDim.x + threadIdx.x; i < n;
         i += gridDim.x * blockDim.x)
        dst[i] = (bf16)src[i];
}

__global__ void copy_f32_kernel(const float* __restrict__ src,
                                float* __restrict__ dst, int n) {
    for (int i = blockIdx.x * blockDim.x + threadIdx.x; i < n;
         i += gridDim.x * blockDim.x)
        dst[i] = src[i];
}

// xs[(s*B + b)*H + h] = bf16(embed[x[b,s], h])   (layout [S,B,H])
__global__ void embed_gather_kernel(const int* __restrict__ x,
                                    const float* __restrict__ embed,
                                    bf16* __restrict__ xs) {
    int i  = blockIdx.x * 256 + threadIdx.x;       // 0 .. S*B*H
    int j  = i % Hc;
    int sb = i / Hc;
    int b  = sb % Bc;
    int s  = sb / Bc;
    int tok = x[b * Sc + s];
    xs[i] = (bf16)embed[(size_t)tok * Hc + j];
}

// ---------------------------------------------------------------- WMMA GEMM
// C[M,N] = A[M,K](bf16) @ W[N,K](bf16)^T  (+ bias[n]), f32 accumulate.
// Block: 256 thr = 8 waves; block tile 64M x 128N; wave tile 16M x 64N.
// K streamed in chunks of 32 through LDS. N % 128 == 0, K % 32 == 0
// (true for all call sites: N in {1024,2048,3072,32000}).
__global__ __launch_bounds__(256) void gemm_bf16_wmma_kernel(
        const bf16* __restrict__ A, int lda,
        const bf16* __restrict__ W, int ldw,
        const float* __restrict__ bias,
        float* __restrict__ C, int ldc,
        int M, int N, int K, int nt_store) {
    __shared__ bf16 sA[64 * 32];
    __shared__ bf16 sW[128 * 32];

    const int tid  = threadIdx.x;
    const int lane = tid & 31;
    const int wave = tid >> 5;
    const int mi   = wave & 3;      // 0..3 : M sub-tile of 16
    const int ni   = wave >> 2;     // 0..1 : N group of 64
    const int m_blk = blockIdx.y * 64;
    const int n_blk = blockIdx.x * 128;

    const int l16  = lane & 15;
    const int half = lane >> 4;     // K half (0: k0..15, 1: k16..31)

    v8f acc[4] = {};

    for (int k0 = 0; k0 < K; k0 += 32) {
        __syncthreads();   // LDS buffers free to overwrite

        // ---- A tile (64 rows x 32 bf16) via Tensor Data Mover.
        // D# 2D tile: tile_dim0=32 elems, tile_dim1=64 rows, stride=lda.
        // tensor_dim1 = M - m_blk: TDM zero-fills OOB rows (M=16 GEMMs).
        if (tid < 32) {
            unsigned lds_base = (unsigned)(size_t)(&sA[0]);
            unsigned long long ga =
                (unsigned long long)(const void*)(A + (size_t)m_blk * lda + k0);
            unsigned td0 = (unsigned)(K - k0);
            unsigned td1 = (unsigned)(M > m_blk ? (M - m_blk) : 1);
            su4 g0;
            g0[0] = 1u;                                            // count=1
            g0[1] = lds_base;                                      // lds_addr
            g0[2] = (unsigned)(ga & 0xFFFFFFFFull);                // gaddr lo
            g0[3] = (unsigned)((ga >> 32) & 0x01FFFFFFull) | (2u << 30); // type=2
            su8 g1;
            g1[0] = 1u << 16;                        // data_size = 2B
            g1[1] = (td0 & 0xFFFFu) << 16;           // tensor_dim0 lo16
            g1[2] = (td0 >> 16) | ((td1 & 0xFFFFu) << 16);  // td0 hi / td1 lo
            g1[3] = (td1 >> 16) | (32u << 16);       // td1 hi / tile_dim0=32
            g1[4] = 64u;                             // tile_dim1=64, tile_dim2=0
            g1[5] = (unsigned)lda;                   // tensor_dim0_stride lo32
            g1[6] = 0u;
            g1[7] = 0u;
            su4 g2 = {0u, 0u, 0u, 0u};
            su4 g3 = {0u, 0u, 0u, 0u};
            asm volatile("tensor_load_to_lds %0, %1, %2, %3"
                         :: "s"(g0), "s"(g1), "s"(g2), "s"(g3)
                         : "memory");
        }

        // ---- W tile (128 rows x 32 bf16): per-lane async copies to LDS.
        #pragma unroll
        for (int it = 0; it < 2; ++it) {
            int idx = tid + it * 256;
            int r   = idx >> 2;
            int cc  = (idx & 3) * 8;
            unsigned lds_off = (unsigned)(size_t)(&sW[r * 32 + cc]);
            const bf16* gp = W + (size_t)(n_blk + r) * ldw + k0 + cc;
            asm volatile("global_load_async_to_lds_b128 %0, %1, off"
                         :: "v"(lds_off), "v"(gp)
                         : "memory");
        }

        // drain own async/tensor counters, then publish to the workgroup
        asm volatile("s_wait_asynccnt 0x0" ::: "memory");
        __builtin_amdgcn_s_wait_tensorcnt(0);
        __syncthreads();

        // prefetch next K-chunk of the weight stream (global_prefetch_b8)
        if (k0 + 32 < K) {
            const bf16* pf = W + (size_t)(n_blk + (tid >> 1)) * ldw +
                             (k0 + 32) + (tid & 1) * 16;
            __builtin_prefetch((const void*)pf, 0, 1);
        }

        // ---- fragments + WMMA
        v16bf af = *(const v16bf*)&sA[(mi * 16 + l16) * 32 + half * 16];
        #pragma unroll
        for (int sub = 0; sub < 4; ++sub) {
            v16bf wf =
                *(const v16bf*)&sW[(ni * 64 + sub * 16 + l16) * 32 + half * 16];
            acc[sub] = __builtin_amdgcn_wmma_f32_16x16x32_bf16(
                /*neg_a=*/false, af, /*neg_b=*/false, wf,
                /*c_mod=*/(short)0, acc[sub],
                /*reuse_a=*/false, /*reuse_b=*/false);
        }
    }

    // --- epilogue: C layout lane n = lane%16, row m = vgpr + 8*(lane/16)
    const int m_base = m_blk + mi * 16 + half * 8;
    #pragma unroll
    for (int sub = 0; sub < 4; ++sub) {
        int n = n_blk + ni * 64 + sub * 16 + l16;
        float bv = bias ? bias[n] : 0.0f;
        #pragma unroll
        for (int v = 0; v < 8; ++v) {
            int m = m_base + v;
            if (m < M) {
                float val = acc[sub][v] + bv;
                float* cp = &C[(size_t)m * ldc + n];
                if (nt_store)
                    __builtin_nontemporal_store(val, cp);
                else
                    *cp = val;
            }
        }
    }
}

// ---------------------------------------------------------------- GRU steps
// t = 0: z = sig(gxz); h = (1-z)*tanh(gxh)
__global__ void gru_first_kernel(const float* __restrict__ gx,
                                 float* __restrict__ h,
                                 bf16* __restrict__ hbf,
                                 bf16* __restrict__ xs_next,
                                 bf16* __restrict__ y_bs, int s) {
    int i = blockIdx.x * 256 + threadIdx.x;    // 0 .. B*H
    int b = i / Hc, j = i - b * Hc;
    size_t base = (size_t)b * 3 * Hc + j;
    float z  = sigmoidf_(gx[base + Hc]);
    float hc = tanhf(gx[base + 2 * Hc]);
    float hn = (1.0f - z) * hc;
    h[i]   = hn;
    hbf[i] = (bf16)hn;
    if (xs_next) xs_next[i] = (bf16)hn;
    if (y_bs)    y_bs[((size_t)b * Sc + s) * Hc + j] = (bf16)hn;
}

// r = sig(gxr + ghr); z = sig(gxz + ghz); hr = h*r
__global__ void gru_rz_kernel(const float* __restrict__ gx_s,
                              const float* __restrict__ gh,
                              const float* __restrict__ h,
                              float* __restrict__ zbuf,
                              bf16* __restrict__ hr) {
    int i = blockIdx.x * 256 + threadIdx.x;
    int b = i / Hc, j = i - b * Hc;
    size_t base = (size_t)b * 3 * Hc + j;
    float r = sigmoidf_(gx_s[base] + gh[base]);
    float z = sigmoidf_(gx_s[base + Hc] + gh[base + Hc]);
    zbuf[i] = z;
    hr[i]   = (bf16)(h[i] * r);
}

// hc = tanh(gxh + (h*r)@Whh); h = z*h + (1-z)*hc
__global__ void gru_h_kernel(const float* __restrict__ gx_s,
                             const float* __restrict__ ghh,  // gh + 2H, stride 3H
                             const float* __restrict__ zbuf,
                             float* __restrict__ h,
                             bf16* __restrict__ hbf,
                             bf16* __restrict__ xs_next,
                             bf16* __restrict__ y_bs, int s,
                             float* __restrict__ state_out) {
    int i = blockIdx.x * 256 + threadIdx.x;
    int b = i / Hc, j = i - b * Hc;
    size_t base = (size_t)b * 3 * Hc + j;
    float hc = tanhf(gx_s[base + 2 * Hc] + ghh[base]);
    float z  = zbuf[i];
    float hn = z * h[i] + (1.0f - z) * hc;
    h[i]   = hn;
    hbf[i] = (bf16)hn;
    if (xs_next)   xs_next[i] = (bf16)hn;
    if (y_bs)      y_bs[((size_t)b * Sc + s) * Hc + j] = (bf16)hn;
    if (state_out) state_out[i] = hn;
}

// ---------------------------------------------------------------- launcher
extern "C" void kernel_launch(void* const* d_in, const int* in_sizes, int n_in,
                              void* d_out, int out_size, void* d_ws,
                              size_t ws_size, hipStream_t stream) {
    const int*   x     = (const int*)d_in[0];
    const float* embed = (const float*)d_in[1];
    const float* Wxr   = (const float*)d_in[2];
    const float* bxr   = (const float*)d_in[3];
    const float* Whr   = (const float*)d_in[4];
    const float* Wxz   = (const float*)d_in[5];
    const float* bxz   = (const float*)d_in[6];
    const float* Whz   = (const float*)d_in[7];
    const float* Wxh   = (const float*)d_in[8];
    const float* bxh   = (const float*)d_in[9];
    const float* Whh   = (const float*)d_in[10];
    const float* Wdec  = (const float*)d_in[11];
    const float* bdec  = (const float*)d_in[12];
    float* out = (float*)d_out;

    // -------- workspace carve (256B aligned)
    char* p = (char*)d_ws;
    auto alloc = [&](size_t bytes) -> char* {
        char* r = p;
        p += (bytes + 255) & ~(size_t)255;
        return r;
    };
    bf16*  wdec_bf = (bf16*)alloc((size_t)Vc * Hc * sizeof(bf16));
    bf16*  wx_bf   = (bf16*)alloc((size_t)Lc * 3 * Hc * Hc * sizeof(bf16));
    bf16*  wh_bf   = (bf16*)alloc((size_t)Lc * 3 * Hc * Hc * sizeof(bf16));
    bf16*  xs_a    = (bf16*)alloc((size_t)Sc * Bc * Hc * sizeof(bf16));
    bf16*  xs_b    = (bf16*)alloc((size_t)Sc * Bc * Hc * sizeof(bf16));
    bf16*  y_bs    = (bf16*)alloc((size_t)Bc * Sc * Hc * sizeof(bf16));
    float* gx      = (float*)alloc((size_t)Sc * Bc * 3 * Hc * sizeof(float));
    float* gh      = (float*)alloc((size_t)Bc * 3 * Hc * sizeof(float));
    float* zbuf    = (float*)alloc((size_t)Bc * Hc * sizeof(float));
    float* hbuf    = (float*)alloc((size_t)Bc * Hc * sizeof(float));
    bf16*  hbf     = (bf16*)alloc((size_t)Bc * Hc * sizeof(bf16));
    bf16*  hr      = (bf16*)alloc((size_t)Bc * Hc * sizeof(bf16));
    float* biasx   = (float*)alloc((size_t)Lc * 3 * Hc * sizeof(float));

    auto cvt = [&](const float* s, bf16* d, int n) {
        int g = (n + 255) / 256;
        if (g > 4096) g = 4096;
        cvt_f32_bf16_kernel<<<g, 256, 0, stream>>>(s, d, n);
    };
    auto cpy = [&](const float* s, float* d, int n) {
        int g = (n + 255) / 256;
        copy_f32_kernel<<<g, 256, 0, stream>>>(s, d, n);
    };
    auto gemm = [&](const bf16* A, int lda, const bf16* W, int ldw,
                    const float* bias, float* C, int ldc, int M, int N, int K,
                    int nt) {
        dim3 grid(N / 128, (M + 63) / 64);
        gemm_bf16_wmma_kernel<<<grid, 256, 0, stream>>>(A, lda, W, ldw, bias,
                                                        C, ldc, M, N, K, nt);
    };

    // -------- weight conversion / bias packing
    const size_t HH = (size_t)Hc * Hc;
    for (int l = 0; l < Lc; ++l) {
        cvt(Wxr + l * HH, wx_bf + ((size_t)l * 3 + 0) * HH, (int)HH);
        cvt(Wxz + l * HH, wx_bf + ((size_t)l * 3 + 1) * HH, (int)HH);
        cvt(Wxh + l * HH, wx_bf + ((size_t)l * 3 + 2) * HH, (int)HH);
        cvt(Whr + l * HH, wh_bf + ((size_t)l * 3 + 0) * HH, (int)HH);
        cvt(Whz + l * HH, wh_bf + ((size_t)l * 3 + 1) * HH, (int)HH);
        cvt(Whh + l * HH, wh_bf + ((size_t)l * 3 + 2) * HH, (int)HH);
        cpy(bxr + l * Hc, biasx + (size_t)l * 3 * Hc + 0 * Hc, Hc);
        cpy(bxz + l * Hc, biasx + (size_t)l * 3 * Hc + 1 * Hc, Hc);
        cpy(bxh + l * Hc, biasx + (size_t)l * 3 * Hc + 2 * Hc, Hc);
    }
    cvt(Wdec, wdec_bf, Vc * Hc);

    // -------- embedding gather to [S,B,H] bf16
    embed_gather_kernel<<<(Sc * Bc * Hc) / 256, 256, 0, stream>>>(x, embed, xs_a);

    // -------- GRU layers
    const bf16* xs_cur = xs_a;
    for (int l = 0; l < Lc; ++l) {
        const bf16* wx_l = wx_bf + (size_t)l * 3 * HH;
        const bf16* wh_l = wh_bf + (size_t)l * 3 * HH;
        bf16* xs_next = (l + 1 < Lc) ? xs_b : nullptr;
        bf16* ybs     = (l == Lc - 1) ? y_bs : nullptr;

        // all-timestep input projection: gx[S*B, 3H]
        gemm(xs_cur, Hc, wx_l, Hc, biasx + (size_t)l * 3 * Hc,
             gx, 3 * Hc, Sc * Bc, 3 * Hc, Hc, /*nt=*/0);

        // s = 0 (no recurrent terms)
        gru_first_kernel<<<(Bc * Hc) / 256, 256, 0, stream>>>(
            gx, hbuf, hbf, xs_next, ybs, 0);

        for (int s = 1; s < Sc; ++s) {
            const float* gx_s = gx + (size_t)s * Bc * 3 * Hc;
            // gh[:, 0:2H] = h @ [Whr;Whz]^T
            gemm(hbf, Hc, wh_l, Hc, nullptr, gh, 3 * Hc, Bc, 2 * Hc, Hc, 0);
            gru_rz_kernel<<<(Bc * Hc) / 256, 256, 0, stream>>>(
                gx_s, gh, hbuf, zbuf, hr);
            // gh[:, 2H:3H] = (h*r) @ Whh^T
            gemm(hr, Hc, wh_l + 2 * HH, Hc, nullptr, gh + 2 * Hc, 3 * Hc,
                 Bc, Hc, Hc, 0);
            float* st = (s == Sc - 1)
                            ? (out + (size_t)Bc * Sc * Vc + (size_t)l * Bc * Hc)
                            : nullptr;
            gru_h_kernel<<<(Bc * Hc) / 256, 256, 0, stream>>>(
                gx_s, gh + 2 * Hc, zbuf, hbuf, hbf,
                xs_next ? xs_next + (size_t)s * Bc * Hc : nullptr, ybs, s, st);
        }
        xs_cur = xs_b;
    }

    // -------- decoder: logits[B*S, V] = y @ Wdec^T + bdec  (NT stores)
    gemm(y_bs, Hc, wdec_bf, Hc, bdec, out, Vc, Sc * Bc, Vc, Hc, /*nt=*/1);
}
